// GroupedQueryAttention_197568495868
// MI455X (gfx1250) — compile-verified, matrix operand order checked by reading, not run on hardware
//
#include <hip/hip_runtime.h>
#include <hip/hip_bf16.h>
#include <stdint.h>

// ---------------------------------------------------------------------------
// GQA forward for MI455X (gfx1250, wave32, WMMA).
// B=2, SQ=SKV=2048, DIN=2048, QH=16, KVH=4, HD=128  (REP=4 along FEATURE axis)
// ---------------------------------------------------------------------------

typedef __attribute__((ext_vector_type(16))) __bf16 v16bf;
typedef __attribute__((ext_vector_type(8)))  float  v8f;

#define B_    2
#define SQ_   2048
#define SKV_  2048
#define DIN_  2048
#define QH_   16
#define HD_   128
#define KVD_  512   // KV_HEADS * HEAD_DIM

__device__ __forceinline__ __bf16 f2bf(float x) { return (__bf16)x; }

// 16x32 bf16 WMMA operand fragment, loaded row-major:
// lane L (<16) -> row L, k-chunk [0..15]; lane L+16 -> row L, k-chunk [16..31].
// Both operands of every product here are loaded with this same pattern from
// row-major [row][k] storage, so any hardware K-slot permutation cancels.
__device__ __forceinline__ v16bf frag_ld(const __bf16* base, int ldm, int lane) {
  return *(const v16bf*)(base + (size_t)(lane & 15) * ldm + ((lane >> 4) << 4));
}

__device__ __forceinline__ v8f wmma_bf16(v16bf a, v16bf b, v8f c) {
  return __builtin_amdgcn_wmma_f32_16x16x32_bf16(false, a, false, b,
                                                 (short)0, c, false, false);
}

// ------------------------------- elementwise -------------------------------

__global__ void __launch_bounds__(256)
cvt_bf16_kernel(const float* __restrict__ in, __bf16* __restrict__ out, int n) {
  int i = blockIdx.x * 256 + threadIdx.x;
  if (i < n) out[i] = f2bf(in[i]);
}

// out[n][k] = (bf16) in[k][n]   (weights stored transposed for symmetric loads)
__global__ void __launch_bounds__(256)
transpose_bf16_kernel(const float* __restrict__ in, __bf16* __restrict__ out,
                      int K, int N) {
  int i = blockIdx.x * 256 + threadIdx.x;
  if (i < K * N) {
    int k = i / N, n = i - k * N;
    out[(size_t)n * K + k] = f2bf(in[i]);
  }
}

// k_exp[bh][s][d] = xk[b, s, h*32 + d/4]   (feature-axis repeat, then heads)
__global__ void __launch_bounds__(256)
expand_k_kernel(const __bf16* __restrict__ xk, __bf16* __restrict__ kexp) {
  int i = blockIdx.x * 256 + threadIdx.x;       // 32*2048*128 = 8388608
  int d = i & 127; int t = i >> 7; int s = t & 2047; int bh = t >> 11;
  int b = bh >> 4, h = bh & 15;
  kexp[i] = xk[((size_t)(b * SKV_ + s)) * KVD_ + h * 32 + (d >> 2)];
}

// vT[bh][d][s] = xv[b, s, h*32 + d/4]   (transposed so PV WMMA loads rows)
__global__ void __launch_bounds__(256)
expand_vT_kernel(const __bf16* __restrict__ xv, __bf16* __restrict__ vT) {
  int i = blockIdx.x * 256 + threadIdx.x;       // 32*128*2048
  int s = i & 2047; int t = i >> 11; int d = t & 127; int bh = t >> 7;
  int b = bh >> 4, h = bh & 15;
  vT[i] = xv[((size_t)(b * SKV_ + s)) * KVD_ + h * 32 + (d >> 2)];
}

// --------------------------------- GEMM ------------------------------------
// C[M][N] = A[M][K] * Bt[N][K]^T, bf16 in, f32 accum; 8 waves, block 128x128,
// wave tile 32x64 (2x4 of 16x16), k-step 32.
__global__ void __launch_bounds__(256)
gemm_bf16_kernel(const __bf16* __restrict__ A, const __bf16* __restrict__ Bt,
                 void* __restrict__ Cout, int M, int N, int K, int out_f32) {
  const int lane = threadIdx.x & 31;
  const int wave = threadIdx.x >> 5;
  const int wm = wave >> 1;                 // 0..3
  const int wn = wave & 1;                  // 0..1
  const int m0 = blockIdx.y * 128 + wm * 32;
  const int n0 = blockIdx.x * 128 + wn * 64;

  v8f acc[2][4] = {};
  const __bf16* a0p = A + (size_t)m0 * K;
  const __bf16* a1p = A + (size_t)(m0 + 16) * K;

  for (int k = 0; k < K; k += 32) {
    __builtin_prefetch(a0p + k + 256, 0, 1);
    v16bf a0 = frag_ld(a0p + k, K, lane);
    v16bf a1 = frag_ld(a1p + k, K, lane);
#pragma unroll
    for (int j = 0; j < 4; ++j) {
      v16bf b = frag_ld(Bt + (size_t)(n0 + j * 16) * K + k, K, lane);
      acc[0][j] = wmma_bf16(a0, b, acc[0][j]);
      acc[1][j] = wmma_bf16(a1, b, acc[1][j]);
    }
  }

  const int rbase = (lane >> 4) << 3;       // C layout: rows r / r+8 per half
  const int cl    = lane & 15;              // N across lanes 0..15
  if (out_f32) {
    float* C = (float*)Cout;
#pragma unroll
    for (int i = 0; i < 2; ++i)
#pragma unroll
      for (int j = 0; j < 4; ++j)
#pragma unroll
        for (int r = 0; r < 8; ++r) {
          int row = m0 + i * 16 + rbase + r;
          int col = n0 + j * 16 + cl;
          if (row < M && col < N) C[(size_t)row * N + col] = acc[i][j][r];
        }
  } else {
    __bf16* C = (__bf16*)Cout;
#pragma unroll
    for (int i = 0; i < 2; ++i)
#pragma unroll
      for (int j = 0; j < 4; ++j)
#pragma unroll
        for (int r = 0; r < 8; ++r) {
          int row = m0 + i * 16 + rbase + r;
          int col = n0 + j * 16 + cl;
          if (row < M && col < N) C[(size_t)row * N + col] = f2bf(acc[i][j][r]);
        }
  }
}

// ------------------------------ attention ----------------------------------
// One block = (b, h, 64-row Q tile); 4 waves, each wave owns 16 Q rows.
// Flash-style streaming over KV in 32-wide tiles staged in LDS.
__global__ void __launch_bounds__(128)
attn_kernel(const __bf16* __restrict__ xq,    // [B*SQ][QH*HD]
            const __bf16* __restrict__ kexp,  // [B*QH][SKV][HD]
            const __bf16* __restrict__ vT,    // [B*QH][HD][SKV]
            const float*  __restrict__ mask,  // [SQ][SKV] (broadcast b,h)
            __bf16* __restrict__ ctx) {       // [B*SQ][QH*HD]
  __shared__ __bf16 sK[32 * 128];             // K tile  [kv 32][d 128]
  __shared__ __bf16 sV[128 * 32];             // Vt tile [d 128][kv 32]
  __shared__ __bf16 sP[4][16 * 32];           // per-wave P round-trip

  const int tid  = threadIdx.x;
  const int lane = tid & 31;
  const int wave = tid >> 5;
  const int qt = blockIdx.x & 31;             // SQ/64 = 32 tiles
  const int bh = blockIdx.x >> 5;             // 0..31
  const int b = bh >> 4, h = bh & 15;
  const int q0 = qt * 64 + wave * 16;

  // Q fragments for this wave's 16 rows, d in 4 chunks of 32
  const __bf16* qptr = xq + ((size_t)(b * SQ_ + q0)) * (QH_ * HD_) + h * HD_;
  v16bf qf[4];
#pragma unroll
  for (int d = 0; d < 4; ++d) qf[d] = frag_ld(qptr + d * 32, QH_ * HD_, lane);

  v8f o[8] = {};
  float mrow[8], lrow[8];
#pragma unroll
  for (int r = 0; r < 8; ++r) { mrow[r] = -1e30f; lrow[r] = 0.0f; }

  const __bf16* kb = kexp + (size_t)bh * SKV_ * HD_;
  const __bf16* vb = vT   + (size_t)bh * HD_ * SKV_;
  __bf16* sPw = sP[wave];
  const float scl = 0.088388347648318447f;    // 1/sqrt(128)

  for (int kv0 = 0; kv0 < SKV_; kv0 += 32) {
    __syncthreads();                          // previous tile fully consumed
    {                                         // cooperative staging, b128 each
      const uint4* ksrc = (const uint4*)(kb + (size_t)kv0 * HD_);  // 8KB contig
      uint4* kdst = (uint4*)sK;
#pragma unroll
      for (int i = 0; i < 4; ++i) kdst[tid + i * 128] = ksrc[tid + i * 128];
      const uint4* vsrc = (const uint4*)(vb + (size_t)tid * SKV_ + kv0);
      uint4* vdst = (uint4*)(sV + tid * 32);
#pragma unroll
      for (int i = 0; i < 4; ++i) vdst[i] = vsrc[i];
    }
    __syncthreads();

    // scores: two 16x16 subtiles covering 32 KV positions
    v8f s0 = {}, s1 = {};
#pragma unroll
    for (int d = 0; d < 4; ++d) {
      v16bf k0 = frag_ld(sK + d * 32, 128, lane);
      v16bf k1 = frag_ld(sK + 16 * 128 + d * 32, 128, lane);
      s0 = wmma_bf16(qf[d], k0, s0);
      s1 = wmma_bf16(qf[d], k1, s1);
    }

    const int rbase = (lane >> 4) << 3;
    const int cl    = lane & 15;
#pragma unroll
    for (int r = 0; r < 8; ++r) {
      const int qi = q0 + rbase + r;
      const float* mp = mask + (size_t)qi * SKV_ + kv0 + cl;
      float x0 = s0[r] * scl - (1.0f / mp[0]  - 1.0f);
      float x1 = s1[r] * scl - (1.0f / mp[16] - 1.0f);
      float mx = fmaxf(x0, x1);
#pragma unroll
      for (int off = 1; off < 16; off <<= 1)
        mx = fmaxf(mx, __shfl_xor(mx, off, 32));
      const float mn = fmaxf(mrow[r], mx);
      const float c  = __expf(mrow[r] - mn);
      const float p0 = __expf(x0 - mn);
      const float p1 = __expf(x1 - mn);
      float rs = p0 + p1;
#pragma unroll
      for (int off = 1; off < 16; off <<= 1) rs += __shfl_xor(rs, off, 32);
      lrow[r] = lrow[r] * c + rs;
      mrow[r] = mn;
#pragma unroll
      for (int dc = 0; dc < 8; ++dc) o[dc][r] *= c;
      sPw[(rbase + r) * 32 + cl]      = f2bf(p0);
      sPw[(rbase + r) * 32 + 16 + cl] = f2bf(p1);
    }

    // P (16x32) * Vt-rows -> 8 output d-chunks of 16
    v16bf pf = frag_ld(sPw, 32, lane);
#pragma unroll
    for (int dc = 0; dc < 8; ++dc) {
      v16bf vf = frag_ld(sV + dc * 16 * 32, 32, lane);
      o[dc] = wmma_bf16(pf, vf, o[dc]);
    }
  }

  const int rbase = (lane >> 4) << 3;
  const int cl    = lane & 15;
#pragma unroll
  for (int dc = 0; dc < 8; ++dc)
#pragma unroll
    for (int r = 0; r < 8; ++r) {
      const int qi = q0 + rbase + r;
      ctx[((size_t)(b * SQ_ + qi)) * (QH_ * HD_) + h * HD_ + dc * 16 + cl] =
          f2bf(o[dc][r] / lrow[r]);
    }
}

// ------------------------------- launcher ----------------------------------

extern "C" void kernel_launch(void* const* d_in, const int* in_sizes, int n_in,
                              void* d_out, int out_size, void* d_ws,
                              size_t ws_size, hipStream_t stream) {
  (void)in_sizes; (void)n_in; (void)out_size; (void)ws_size;
  const float* q    = (const float*)d_in[0];
  const float* kv   = (const float*)d_in[1];
  const float* mask = (const float*)d_in[2];
  const float* Wq   = (const float*)d_in[3];
  const float* Wk   = (const float*)d_in[4];
  const float* Wv   = (const float*)d_in[5];
  const float* Wo   = (const float*)d_in[6];

  char* ws = (char*)d_ws;                     // ~124 MB total scratch
  __bf16* qb   = (__bf16*)(ws + 0);           // 16 MB  [4096][2048]
  __bf16* kvb  = (__bf16*)(ws + 16777216);    // 16 MB  [4096][2048]
  __bf16* WqT  = (__bf16*)(ws + 33554432);    //  8 MB  [2048][2048]
  __bf16* WkT  = (__bf16*)(ws + 41943040);    //  2 MB  [512][2048]
  __bf16* WvT  = (__bf16*)(ws + 44040192);    //  2 MB  [512][2048]
  __bf16* WoT  = (__bf16*)(ws + 46137344);    //  8 MB  [2048][2048]
  __bf16* xq   = (__bf16*)(ws + 54525952);    // 16 MB  [4096][2048]
  __bf16* xk   = (__bf16*)(ws + 71303168);    //  4 MB  [4096][512]
  __bf16* xv   = (__bf16*)(ws + 75497472);    //  4 MB  [4096][512]
  __bf16* kexp = (__bf16*)(ws + 79691776);    // 16 MB  [32][2048][128]
  __bf16* vTx  = (__bf16*)(ws + 96468992);    // 16 MB  [32][128][2048]
  __bf16* ctx  = (__bf16*)(ws + 113246208);   // 16 MB  [4096][2048]

  cvt_bf16_kernel<<<8388608 / 256, 256, 0, stream>>>(q, qb, 8388608);
  cvt_bf16_kernel<<<8388608 / 256, 256, 0, stream>>>(kv, kvb, 8388608);
  transpose_bf16_kernel<<<(2048 * 2048) / 256, 256, 0, stream>>>(Wq, WqT, 2048, 2048);
  transpose_bf16_kernel<<<(2048 * 512) / 256, 256, 0, stream>>>(Wk, WkT, 2048, 512);
  transpose_bf16_kernel<<<(2048 * 512) / 256, 256, 0, stream>>>(Wv, WvT, 2048, 512);
  transpose_bf16_kernel<<<(2048 * 2048) / 256, 256, 0, stream>>>(Wo, WoT, 2048, 2048);

  { dim3 g(2048 / 128, 4096 / 128);
    gemm_bf16_kernel<<<g, 256, 0, stream>>>(qb, WqT, xq, 4096, 2048, 2048, 0); }
  { dim3 g(512 / 128, 4096 / 128);
    gemm_bf16_kernel<<<g, 256, 0, stream>>>(kvb, WkT, xk, 4096, 512, 2048, 0); }
  { dim3 g(512 / 128, 4096 / 128);
    gemm_bf16_kernel<<<g, 256, 0, stream>>>(kvb, WvT, xv, 4096, 512, 2048, 0); }

  expand_k_kernel<<<8388608 / 256, 256, 0, stream>>>(xk, kexp);
  expand_vT_kernel<<<8388608 / 256, 256, 0, stream>>>(xv, vTx);

  attn_kernel<<<1024, 128, 0, stream>>>(xq, kexp, vTx, mask, ctx);

  { dim3 g(2048 / 128, 4096 / 128);
    gemm_bf16_kernel<<<g, 256, 0, stream>>>(ctx, WoT, d_out, 4096, 2048, 2048, 1); }
}
